// BEVNet_180388627083
// MI455X (gfx1250) — matrix operand order repaired and checked by reading, not compiled, not run on hardware
//
#include <hip/hip_runtime.h>
#include <hip/hip_bf16.h>
#include <cstddef>

// ---------------------------------------------------------------------------
// Types for CDNA5 WMMA (wave32, 16x16x32 bf16 -> f32)
// ---------------------------------------------------------------------------
typedef __attribute__((ext_vector_type(16))) __bf16          v16bf;
typedef __attribute__((ext_vector_type(8)))  float           v8f;
typedef __attribute__((ext_vector_type(8)))  unsigned short  us8;   // 16B, for b128 LDS/global ops
typedef int g4i __attribute__((vector_size(16)));                   // matches async builtin pointee

union Frag { us8 h[2]; v16bf v; };

#define AS1 __attribute__((address_space(1)))
#define AS3 __attribute__((address_space(3)))

#if defined(__gfx1250__) && __has_builtin(__builtin_amdgcn_global_load_async_to_lds_b128)
#define HAVE_ASYNC_LDS 1
#endif

#ifdef HAVE_ASYNC_LDS
#if __has_builtin(__builtin_amdgcn_s_wait_asynccnt)
#define WAIT_ASYNC() __builtin_amdgcn_s_wait_asynccnt(0)
#else
#define WAIT_ASYNC() asm volatile("s_wait_asynccnt 0" ::: "memory")
#endif
#else
#define WAIT_ASYNC()
#endif

__device__ __forceinline__ unsigned short f2bf(float f) {
    unsigned int x = __float_as_uint(f);
    unsigned int r = (x + 0x7FFFu + ((x >> 16) & 1u)) >> 16;   // RNE truncation
    return (unsigned short)r;
}
__device__ __forceinline__ float bf2f(unsigned short u) {
    return __uint_as_float(((unsigned int)u) << 16);
}

// ---------------------------------------------------------------------------
// Implicit-GEMM 3x3 SAME conv, stride 1/2, bf16 WMMA, fused BN(+residual)+ReLU
//   M = OH*OW, N = Cout, K = Cin*9 (weights pre-padded to Kpad x Coutp)
//   block = 128 threads = 4 waves in a 2x2 grid of 16x16 C tiles
//   LDS rows padded to 40 ushorts (80B: 16B-aligned, 20-bank stride)
// ---------------------------------------------------------------------------
#define TM 32
#define TN 32
#define TK 32
#define LPAD 8   // row stride TK+LPAD = 40 ushorts

__global__ __launch_bounds__(128)
void k_conv3x3_wmma(const void* __restrict__ src, int srcF32,
                    const unsigned short* __restrict__ wKN,   // bf16 [Kpad][Coutp]
                    const float* __restrict__ scale,
                    const float* __restrict__ bias, int hasScale,
                    unsigned short* __restrict__ dst,         // bf16 [Cout][M]
                    const unsigned short* __restrict__ residual,
                    int Cin, int Cout, int Coutp, int H, int W, int stride, int relu)
{
    __shared__ __align__(16) unsigned short As[2][TM][TK + LPAD];  // [pixel][k]
    __shared__ __align__(16) unsigned short Bs[2][TN][TK + LPAD];  // [cout][k] (transposed)
#ifdef HAVE_ASYNC_LDS
    __shared__ __align__(16) unsigned short Bst[2][TK][TN];        // async staging [k][cout]
#endif

    const int OH = H / stride, OW = W / stride;
    const int M = OH * OW;
    const int K = Cin * 9;
    const int tM = blockIdx.x * TM;
    const int tN = blockIdx.y * TN;
    const int t = threadIdx.x;
    const int lane = t & 31;          // wave32
    const int wv = t >> 5;
    const int wm = wv & 1, wn = wv >> 1;
    const int half = lane >> 4, lr = lane & 15;
    const int klB = t >> 2, segB = t & 3;   // B-tile loader: 16B per thread

    const float*          srcF = (const float*)src;
    const unsigned short* srcH = (const unsigned short*)src;

    auto issueTiles = [&](int buf, int kk) {
        // ---- A tile: 32 pixels x 32 k, im2col gather (8 elements/thread) ----
        for (int i = t; i < TM * TK; i += 128) {
            int row = i >> 5, kl = i & 31;
            int k = kk + kl;
            int p = tM + row;
            unsigned short v = 0;
            if (k < K && p < M) {
                int ci = k / 9, r9 = k - ci * 9;
                int kh = r9 / 3, kw = r9 - kh * 3;
                int oh = p / OW, ow = p - oh * OW;
                int ih = oh * stride + kh - 1;
                int iw = ow * stride + kw - 1;
                if ((unsigned)ih < (unsigned)H && (unsigned)iw < (unsigned)W) {
                    size_t idx = (size_t)ci * H * W + (size_t)ih * W + iw;
                    v = srcF32 ? f2bf(srcF[idx]) : srcH[idx];
                }
            }
            As[buf][row][kl] = v;
        }
        // ---- B tile: 32 k x 32 cout (zero-padded weights -> no bounds) ----
        {
            const unsigned short* gp = wKN + (size_t)(kk + klB) * Coutp + tN + segB * 8;
#ifdef HAVE_ASYNC_LDS
            // CDNA5 async copy straight into LDS staging, tracked on ASYNCcnt
            __builtin_amdgcn_global_load_async_to_lds_b128(
                (AS1 g4i*)gp,
                (AS3 g4i*)&Bst[buf][klB][segB * 8], 0, 0);
#else
            us8 v = *(const us8*)gp;
#pragma unroll
            for (int j = 0; j < 8; ++j) Bs[buf][segB * 8 + j][klB] = v[j];
#endif
        }
        if (t == 0 && (kk + 2 * TK) < K)
            __builtin_prefetch(wKN + (size_t)(kk + 2 * TK) * Coutp + tN, 0, 1);
    };

#ifdef HAVE_ASYNC_LDS
    auto transposeB = [&](int buf) {
        us8 v = *(const us8*)&Bst[buf][klB][segB * 8];    // ds_load_b128
#pragma unroll
        for (int j = 0; j < 8; ++j) Bs[buf][segB * 8 + j][klB] = v[j];
    };
#endif

    v8f acc = {0.f, 0.f, 0.f, 0.f, 0.f, 0.f, 0.f, 0.f};

    const int ktiles = (K + TK - 1) / TK;
    issueTiles(0, 0);
    WAIT_ASYNC();
    __syncthreads();

    for (int kt = 0; kt < ktiles; ++kt) {
        int cur = kt & 1;
        if (kt + 1 < ktiles) issueTiles(cur ^ 1, (kt + 1) * TK);
#ifdef HAVE_ASYNC_LDS
        transposeB(cur);
        __syncthreads();           // Bs[cur] visible to all waves
#endif
        // Fragments: two contiguous 16B chunks per operand (ds_load_b128).
        //  A (16x32 bf16): lanes 0-15 hold K 0..7 / 16..23, lanes 16-31 K 8..15 / 24..31.
        //  B: per-lane column, same K order as A (lanes 0-15 K 0..7 / 16..23, lanes 16-31 K 8..15 / 24..31).
        Frag a, b;
        a.h[0] = *(const us8*)&As[cur][wm * 16 + lr][8 * half];
        a.h[1] = *(const us8*)&As[cur][wm * 16 + lr][16 + 8 * half];
        b.h[0] = *(const us8*)&Bs[cur][wn * 16 + lr][8 * half];
        b.h[1] = *(const us8*)&Bs[cur][wn * 16 + lr][16 + 8 * half];

        acc = __builtin_amdgcn_wmma_f32_16x16x32_bf16(
            false, a.v, false, b.v, (short)0, acc, false, false);

        WAIT_ASYNC();              // next-buffer staging done (overlapped with wmma)
        __syncthreads();
    }

    // Epilogue: C/D layout -> VGPR r: lanes0-15 M=r, lanes16-31 M=r+8.
    // Each lane owns 8 consecutive pixels of one channel -> single b128 store.
    int co = tN + wn * 16 + lr;
    int pbase = tM + wm * 16 + 8 * half;
    if (co < Cout) {
        float s = hasScale ? scale[co] : 1.0f;
        float bb = bias ? bias[co] : 0.0f;
        if (pbase + 8 <= M) {
            us8 o;
            if (residual) {
                us8 rsd = *(const us8*)(residual + (size_t)co * M + pbase);
#pragma unroll
                for (int r = 0; r < 8; ++r) {
                    float v = acc[r] * s + bb + bf2f(rsd[r]);
                    if (relu && v < 0.f) v = 0.f;
                    o[r] = f2bf(v);
                }
            } else {
#pragma unroll
                for (int r = 0; r < 8; ++r) {
                    float v = acc[r] * s + bb;
                    if (relu && v < 0.f) v = 0.f;
                    o[r] = f2bf(v);
                }
            }
            *(us8*)(dst + (size_t)co * M + pbase) = o;
        } else {
#pragma unroll
            for (int r = 0; r < 8; ++r) {
                int p = pbase + r;
                if (p < M) {
                    float v = acc[r] * s + bb;
                    if (residual) v += bf2f(residual[(size_t)co * M + p]);
                    if (relu && v < 0.f) v = 0.f;
                    dst[(size_t)co * M + p] = f2bf(v);
                }
            }
        }
    }
}

// ---------------------------------------------------------------------------
// Helper kernels
// ---------------------------------------------------------------------------
// OIHW f32 -> [Kpad][Coutp] bf16, zero padded (K = Cin*9)
__global__ void k_wtrans(const float* __restrict__ w, unsigned short* __restrict__ o,
                         int Cout, int K, int Coutp, size_t total) {
    size_t i = (size_t)blockIdx.x * blockDim.x + threadIdx.x;
    if (i >= total) return;
    int k = (int)(i / Coutp);
    int co = (int)(i - (size_t)k * Coutp);
    unsigned short v = 0;
    if (k < K && co < Cout) v = f2bf(w[(size_t)co * K + k]);
    o[i] = v;
}

__global__ void k_fill0(float* p, size_t n) {
    size_t i = (size_t)blockIdx.x * blockDim.x + threadIdx.x;
    if (i < n) p[i] = 0.f;
}

__global__ void k_f32bf(const float* __restrict__ s, unsigned short* __restrict__ d, size_t n) {
    size_t i = (size_t)blockIdx.x * blockDim.x + threadIdx.x;
    if (i < n) d[i] = f2bf(s[i]);
}

__global__ void k_bilinear(const unsigned short* __restrict__ feat,
                           const float* __restrict__ coord,
                           float* __restrict__ out,
                           int C, int H, int W, int N, float sc) {
    size_t total = (size_t)C * N;
    size_t i = (size_t)blockIdx.x * blockDim.x + threadIdx.x;
    if (i >= total) return;
    int c = (int)(i / N);
    int n = (int)(i - (size_t)c * N);
    float h = coord[2 * n + 0] * sc;
    float w = coord[2 * n + 1] * sc;
    float h0 = floorf(h), w0 = floorf(w);
    float dh = h - h0, dw = w - w0;
    int h0i = (int)h0, w0i = (int)w0;
    auto g = [&](int hi, int wi) -> float {
        bool valid = (hi >= 0) && (hi < H) && (wi >= 0) && (wi < W);
        int hc = hi < 0 ? 0 : (hi > H - 1 ? H - 1 : hi);
        int wc = wi < 0 ? 0 : (wi > W - 1 ? W - 1 : wi);
        float v = bf2f(feat[(size_t)c * H * W + (size_t)hc * W + wc]);
        return valid ? v : 0.f;
    };
    out[i] = g(h0i, w0i) * (1.f - dh) * (1.f - dw)
           + g(h0i, w0i + 1) * (1.f - dh) * dw
           + g(h0i + 1, w0i) * dh * (1.f - dw)
           + g(h0i + 1, w0i + 1) * dh * dw;
}

// max-pool scatter; features are non-negative (post-ReLU), so int atomicMax
// on the IEEE bit pattern is order-preserving.
__global__ void k_voxmax(const float* __restrict__ pts, const float* __restrict__ coord,
                         float* __restrict__ out, int C, int H, int W, int N, float sc) {
    size_t total = (size_t)C * N;
    size_t i = (size_t)blockIdx.x * blockDim.x + threadIdx.x;
    if (i >= total) return;
    int c = (int)(i / N);
    int n = (int)(i - (size_t)c * N);
    int hi = (int)floorf(coord[2 * n + 0] * sc);
    int wi = (int)floorf(coord[2 * n + 1] * sc);
    if (hi >= 0 && hi < H && wi >= 0 && wi < W) {
        float v = pts[i];
        if (v < 0.f) v = 0.f;
        atomicMax((int*)&out[(size_t)c * H * W + (size_t)hi * W + wi], __float_as_int(v));
    }
}

__global__ void k_resize(const unsigned short* __restrict__ src, unsigned short* __restrict__ dst,
                         int C, int h, int w, int H, int W) {
    size_t total = (size_t)C * H * W;
    size_t i = (size_t)blockIdx.x * blockDim.x + threadIdx.x;
    if (i >= total) return;
    int c = (int)(i / ((size_t)H * W));
    int rem = (int)(i - (size_t)c * H * W);
    int y = rem / W, x = rem - y * W;
    float sy = ((y + 0.5f) * h) / H - 0.5f;
    float sx = ((x + 0.5f) * w) / W - 0.5f;
    float y0f = floorf(sy), x0f = floorf(sx);
    int y0 = (int)y0f, x0 = (int)x0f;
    float dy = sy - y0f, dx = sx - x0f;
    auto at = [&](int yy, int xx) -> float {
        yy = yy < 0 ? 0 : (yy > h - 1 ? h - 1 : yy);
        xx = xx < 0 ? 0 : (xx > w - 1 ? w - 1 : xx);
        return bf2f(src[(size_t)c * h * w + (size_t)yy * w + xx]);
    };
    float v = at(y0, x0) * (1.f - dy) * (1.f - dx)
            + at(y0, x0 + 1) * (1.f - dy) * dx
            + at(y0 + 1, x0) * dy * (1.f - dx)
            + at(y0 + 1, x0 + 1) * dy * dx;
    dst[i] = f2bf(v);
}

__global__ void k_merge(const unsigned short* __restrict__ lo,
                        const unsigned short* __restrict__ hi,
                        const unsigned short* __restrict__ att,
                        unsigned short* __restrict__ dstH, float* __restrict__ dstF,
                        int C, int M) {
    size_t total = (size_t)C * M;
    size_t i = (size_t)blockIdx.x * blockDim.x + threadIdx.x;
    if (i >= total) return;
    int p = (int)(i % M);
    float a0 = bf2f(att[p]), a1 = bf2f(att[(size_t)M + p]);
    float mx = a0 > a1 ? a0 : a1;
    float e0 = __expf(a0 - mx), e1 = __expf(a1 - mx);
    float w0 = e0 / (e0 + e1);
    float v = bf2f(lo[i]) * w0 + bf2f(hi[i]) * (1.f - w0);
    if (dstH) dstH[i] = f2bf(v);
    if (dstF) dstF[i] = v;
}

// ---------------------------------------------------------------------------
// Host orchestration
// ---------------------------------------------------------------------------
extern "C" void kernel_launch(void* const* d_in, const int* in_sizes, int n_in,
                              void* d_out, int out_size, void* d_ws, size_t ws_size,
                              hipStream_t stream) {
    (void)in_sizes; (void)n_in; (void)out_size; (void)ws_size;
    const int NP = 160000;

    const float* cIn    = (const float*)d_in[0];
    /* d_in[1] = p  (unused by reference forward) */
    const float* ccoord = (const float*)d_in[2];
    const float* pcoord = (const float*)d_in[3];

    int curp = 4;
    auto nxt = [&]() { return (const float*)d_in[curp++]; };
    struct Blk   { const float *w1,*s1,*b1,*w2,*s2,*b2; };
    struct Layer { const float *dsw,*dss,*dsb; Blk b[3]; };
    struct Att   { const float *lw,*ls,*lb,*hw,*hs,*hb,*aw1,*as1,*ab1,*aw2,*abias; };
    auto rdLayer = [&]() {
        Layer L; L.dsw = nxt(); L.dss = nxt(); L.dsb = nxt();
        for (int i = 0; i < 3; i++) {
            L.b[i].w1 = nxt(); L.b[i].s1 = nxt(); L.b[i].b1 = nxt();
            L.b[i].w2 = nxt(); L.b[i].s2 = nxt(); L.b[i].b2 = nxt();
        }
        return L;
    };
    auto rdAtt = [&]() {
        Att A; A.lw = nxt(); A.ls = nxt(); A.lb = nxt();
        A.hw = nxt(); A.hs = nxt(); A.hb = nxt();
        A.aw1 = nxt(); A.as1 = nxt(); A.ab1 = nxt();
        A.aw2 = nxt(); A.abias = nxt();
        return A;
    };
    Layer Lch = rdLayer(), Lph = rdLayer(), Lc1 = rdLayer(), Lp1 = rdLayer(), Lc2 = rdLayer();
    Att Au2 = rdAtt(), Au1 = rdAtt();

    // workspace bump allocator
    size_t off = 0;
    auto alloc = [&](size_t bytes) -> void* {
        void* p = (char*)d_ws + off;
        off = (off + bytes + 255) & ~(size_t)255;
        return p;
    };
    unsigned short* wbuf  = (unsigned short*)alloc((size_t)589824 * 2);        // max Kpad*Coutp
    unsigned short* c0buf = (unsigned short*)alloc((size_t)64  * 65536 * 2);   // (64,256^2)
    unsigned short* c1buf = (unsigned short*)alloc((size_t)128 * 16384 * 2);   // (128,128^2)
    unsigned short* c2buf = (unsigned short*)alloc((size_t)256 * 4096  * 2);   // (256,64^2)
    unsigned short* c3buf = (unsigned short*)alloc((size_t)96  * 16384 * 2);   // (96,128^2)
    unsigned short* t0    = (unsigned short*)alloc((size_t)128 * 65536 * 2);
    unsigned short* t1    = (unsigned short*)alloc((size_t)64  * 65536 * 2);
    unsigned short* t2    = (unsigned short*)alloc((size_t)96  * 65536 * 2);
    float*          pf32  = (float*)alloc((size_t)32 * 512 * 512 * 4);
    float*          ptf   = (float*)alloc((size_t)64 * NP * 4);

    auto gsz = [](size_t n) { return dim3((unsigned)((n + 255) / 256)); };
    auto cdiv = [](int a, int b) { return (a + b - 1) / b; };

    auto conv = [&](const void* src, int f32, const float* w, const float* s, const float* b,
                    int Cin, int Cout, int H, int W, int stride, int relu,
                    const unsigned short* res, unsigned short* dst) {
        int K = Cin * 9;
        int Kpad = 32 * cdiv(K, 32);
        int Coutp = 32 * cdiv(Cout, 32);
        size_t nw = (size_t)Kpad * Coutp;
        k_wtrans<<<gsz(nw), 256, 0, stream>>>(w, wbuf, Cout, K, Coutp, nw);
        int OH = H / stride, OW = W / stride, M = OH * OW;
        dim3 g(cdiv(M, TM), cdiv(Cout, TN));
        k_conv3x3_wmma<<<g, 128, 0, stream>>>(src, f32, wbuf, s, b, s ? 1 : 0,
                                              dst, res, Cin, Cout, Coutp, H, W, stride, relu);
    };

    auto run_layer = [&](const void* src, int f32, int Cin, int H, int W, const Layer& L,
                         int Cout, unsigned short* x, unsigned short* ta, unsigned short* tb) {
        conv(src, f32, L.dsw, L.dss, L.dsb, Cin, Cout, H, W, 2, 1, nullptr, x);
        int oh = H / 2, ow = W / 2;
        // block 0:  x -> tb
        conv(x,  0, L.b[0].w1, L.b[0].s1, L.b[0].b1, Cout, Cout, oh, ow, 1, 1, nullptr, ta);
        conv(ta, 0, L.b[0].w2, L.b[0].s2, L.b[0].b2, Cout, Cout, oh, ow, 1, 1, x,  tb);
        // block 1:  tb -> x
        conv(tb, 0, L.b[1].w1, L.b[1].s1, L.b[1].b1, Cout, Cout, oh, ow, 1, 1, nullptr, ta);
        conv(ta, 0, L.b[1].w2, L.b[1].s2, L.b[1].b2, Cout, Cout, oh, ow, 1, 1, tb, x);
        // block 2:  x -> x (per-element in-place residual is safe)
        conv(x,  0, L.b[2].w1, L.b[2].s1, L.b[2].b1, Cout, Cout, oh, ow, 1, 1, nullptr, ta);
        conv(ta, 0, L.b[2].w2, L.b[2].s2, L.b[2].b2, Cout, Cout, oh, ow, 1, 1, x,  x);
    };

    auto att_merge = [&](const unsigned short* xl, int Cl, const unsigned short* xh, int Ch,
                         int H, int W, const Att& A, int cout,
                         unsigned short* dstH, float* dstF) {
        int M = H * W, hh = H / 2, hw = W / 2;
        k_resize<<<gsz((size_t)Ch * M), 256, 0, stream>>>(xh, t2, Ch, hh, hw, H, W);
        conv(xl, 0, A.lw, A.ls, A.lb, Cl, cout, H, W, 1, 1, nullptr, t0);
        conv(t2, 0, A.hw, A.hs, A.hb, Ch, cout, H, W, 1, 1, nullptr, t0 + (size_t)cout * M);
        int cm = cout / 2;
        conv(t0, 0, A.aw1, A.as1, A.ab1, 2 * cout, cm, H, W, 1, 1, nullptr, t1);
        conv(t1, 0, A.aw2, nullptr, A.abias, cm, 2, H, W, 1, 0, nullptr, t2);
        k_merge<<<gsz((size_t)cout * M), 256, 0, stream>>>(
            t0, t0 + (size_t)cout * M, t2, dstH, dstF, cout, M);
    };

    // ---- forward graph ----
    // cart header: (192,512^2) f32 -> c0a (32,256^2) bf16 at c0buf[:32]
    run_layer(cIn, 1, 192, 512, 512, Lch, 32, c0buf, t0, t1);

    // c0_point -> polar pool (32,512^2)
    k_bilinear<<<gsz((size_t)32 * NP), 256, 0, stream>>>(c0buf, ccoord, ptf, 32, 256, 256, NP, 0.5f);
    k_fill0   <<<gsz((size_t)32 * 512 * 512), 256, 0, stream>>>(pf32, (size_t)32 * 512 * 512);
    k_voxmax  <<<gsz((size_t)32 * NP), 256, 0, stream>>>(ptf, pcoord, pf32, 32, 512, 512, NP, 1.0f);

    // polar header: (32,512^2) f32 -> (32,256^2) bf16 at t2
    run_layer(pf32, 1, 32, 512, 512, Lph, 32, t2, t0, t1);

    // back to cartesian: pool into (32,256^2), append to c0buf -> c0 (64,256^2)
    k_bilinear<<<gsz((size_t)32 * NP), 256, 0, stream>>>(t2, pcoord, ptf, 32, 256, 256, NP, 0.5f);
    k_fill0   <<<gsz((size_t)32 * 256 * 256), 256, 0, stream>>>(pf32, (size_t)32 * 256 * 256);
    k_voxmax  <<<gsz((size_t)32 * NP), 256, 0, stream>>>(ptf, ccoord, pf32, 32, 256, 256, NP, 0.5f);
    k_f32bf   <<<gsz((size_t)32 * 65536), 256, 0, stream>>>(pf32, c0buf + (size_t)32 * 65536,
                                                            (size_t)32 * 65536);

    // cart_res1: c0 (64,256^2) -> c1a (64,128^2) at c1buf[:64]
    run_layer(c0buf, 0, 64, 256, 256, Lc1, 64, c1buf, t0, t1);

    // c1_point_out (output #2, f32) straight into d_out, also feeds polar pool
    float* c1pt = (float*)d_out + (size_t)64 * 65536;
    k_bilinear<<<gsz((size_t)64 * NP), 256, 0, stream>>>(c1buf, ccoord, c1pt, 64, 128, 128, NP, 0.25f);
    k_fill0   <<<gsz((size_t)64 * 256 * 256), 256, 0, stream>>>(pf32, (size_t)64 * 256 * 256);
    k_voxmax  <<<gsz((size_t)64 * NP), 256, 0, stream>>>(c1pt, pcoord, pf32, 64, 256, 256, NP, 0.5f);

    // polar_res1: (64,256^2) f32 -> (64,128^2) bf16 at t2
    run_layer(pf32, 1, 64, 256, 256, Lp1, 64, t2, t0, t1);

    // back to cartesian: (64,128^2), append to c1buf -> c1 (128,128^2)
    k_bilinear<<<gsz((size_t)64 * NP), 256, 0, stream>>>(t2, pcoord, ptf, 64, 128, 128, NP, 0.25f);
    k_fill0   <<<gsz((size_t)64 * 128 * 128), 256, 0, stream>>>(pf32, (size_t)64 * 128 * 128);
    k_voxmax  <<<gsz((size_t)64 * NP), 256, 0, stream>>>(ptf, ccoord, pf32, 64, 128, 128, NP, 0.25f);
    k_f32bf   <<<gsz((size_t)64 * 16384), 256, 0, stream>>>(pf32, c1buf + (size_t)64 * 16384,
                                                            (size_t)64 * 16384);

    // cart_res2: c1 (128,128^2) -> c2 (256,64^2)
    run_layer(c1buf, 0, 128, 128, 128, Lc2, 256, c2buf, t0, t1);

    // up2: merge(c1, c2) -> c3 (96,128^2) bf16
    att_merge(c1buf, 128, c2buf, 256, 128, 128, Au2, 96, c3buf, nullptr);

    // up1: merge(c0, c3) -> c4 (64,256^2) f32 into d_out (output #1)
    att_merge(c0buf, 64, c3buf, 96, 256, 256, Au1, 64, nullptr, (float*)d_out);
}
